// Multihead_self_attention_59459527246028
// MI455X (gfx1250) — matrix operand order repaired and checked by reading, not compiled, not run on hardware
//
#include <hip/hip_runtime.h>

// ---------- types ----------
typedef unsigned short u16;
typedef unsigned int   u32;
typedef __bf16 v16bf __attribute__((ext_vector_type(16)));
typedef float  v8f   __attribute__((ext_vector_type(8)));
typedef u32    v4u   __attribute__((ext_vector_type(4)));

union Frag16 { v16bf v; v4u q[2]; };   // 32B: one WMMA 16-bit A/B operand

// ---------- CDNA5 async global->LDS path (guarded; falls back to reg staging) ----------
#if defined(__gfx1250__) && __has_builtin(__builtin_amdgcn_global_load_async_to_lds_b128)
#define ATH_ASYNC 1
#else
#define ATH_ASYNC 0
#endif

#if ATH_ASYNC
typedef int v4i_vs __attribute__((vector_size(16)));
typedef __attribute__((address_space(1))) v4i_vs* as1_v4i;   // global
typedef __attribute__((address_space(3))) v4i_vs* as3_v4i;   // LDS
#endif

__device__ __forceinline__ void cp16_g2l(u16* lds, const u16* g) {
#if ATH_ASYNC
  __builtin_amdgcn_global_load_async_to_lds_b128((as1_v4i)g, (as3_v4i)lds, 0, 0);
#else
  *(v4u*)lds = *(const v4u*)g;
#endif
}

template <int N>
__device__ __forceinline__ void wait_async() {
#if ATH_ASYNC
#if __has_builtin(__builtin_amdgcn_s_wait_asynccnt)
  __builtin_amdgcn_s_wait_asynccnt((unsigned short)N);
#else
  asm volatile("s_wait_asynccnt %0" ::"i"(N) : "memory");
#endif
#endif
}

// ---------- problem constants ----------
constexpr int Bc  = 2;
constexpr int Sc  = 2048;
constexpr int Dc  = 1024;
constexpr int Hc  = 16;
constexpr int DHc = 64;
constexpr int Mtot = Bc * Sc;          // 4096 rows

// ---------- workspace layout (bytes) ----------
constexpr size_t OFF_XB = 0;                                   // x   bf16 [4096,1024]
constexpr size_t OFF_WQ = OFF_XB + (size_t)Mtot * Dc * 2;      // WQ  bf16 [1024,1024]
constexpr size_t OFF_WK = OFF_WQ + (size_t)Dc * Dc * 2;
constexpr size_t OFF_WV = OFF_WK + (size_t)Dc * Dc * 2;
constexpr size_t OFF_WO = OFF_WV + (size_t)Dc * Dc * 2;
constexpr size_t OFF_QF = OFF_WO + (size_t)Dc * Dc * 2;        // Q   f32  [4096,1024]
constexpr size_t OFF_KF = OFF_QF + (size_t)Mtot * Dc * 4;
constexpr size_t OFF_VF = OFF_KF + (size_t)Mtot * Dc * 4;
constexpr size_t OFF_QB = OFF_VF + (size_t)Mtot * Dc * 4;      // Q   bf16 [b,h,s,d]
constexpr size_t OFF_KB = OFF_QB + (size_t)Mtot * Dc * 2;      // K   bf16 [b,h,s,d]
constexpr size_t OFF_VT = OFF_KB + (size_t)Mtot * Dc * 2;      // V^T bf16 [b,h,d,s]
constexpr size_t OFF_AB = OFF_VT + (size_t)Mtot * Dc * 2;      // attn out bf16 [4096,1024]
// total = 96 MB

// ---------- helpers ----------
__device__ __forceinline__ u16 f2b(float f) {        // f32 -> bf16, round-nearest-even
  u32 x = __builtin_bit_cast(u32, f);
  x += 0x7FFFu + ((x >> 16) & 1u);
  return (u16)(x >> 16);
}

__device__ __forceinline__ v8f wmma_bf16(v16bf a, v16bf b, v8f c) {
  // D = A(16x32 bf16) * B(32x16 bf16) + C(16x16 f32)
  return __builtin_amdgcn_wmma_f32_16x16x32_bf16(false, a, false, b, (short)0, c,
                                                 false, false);
}

__device__ __forceinline__ float half_max(float v) { // reduce over 16-lane half
  v = fmaxf(v, __shfl_xor(v, 1, 32));
  v = fmaxf(v, __shfl_xor(v, 2, 32));
  v = fmaxf(v, __shfl_xor(v, 4, 32));
  v = fmaxf(v, __shfl_xor(v, 8, 32));
  return v;
}
__device__ __forceinline__ float half_sum(float v) {
  v += __shfl_xor(v, 1, 32);
  v += __shfl_xor(v, 2, 32);
  v += __shfl_xor(v, 4, 32);
  v += __shfl_xor(v, 8, 32);
  return v;
}

// ---------- elementwise f32 -> bf16 ----------
__global__ __launch_bounds__(256) void k_cvt(const float* __restrict__ src,
                                             u16* __restrict__ dst, int n) {
  int i = blockIdx.x * 256 + threadIdx.x;
  if (i < n) dst[i] = f2b(src[i]);
}

// ---------- NT GEMM: C[m,n] = sum_k A[m,k] * W[n,k] ----------
// A: [M,K] bf16 row-major, W: [N,K] bf16 row-major, C: [M,N] f32.
// Block tile 128x256, k-step 32, 8 waves (2 in M x 4 in N), wave tile 64x64.
// Double-buffered LDS fed by async global->LDS copies (ASYNCcnt pipeline).
constexpr int LDT = 40;                 // LDS row stride (halves), pad vs 64 banks
__global__ __launch_bounds__(256) void k_gemm_nt(const u16* __restrict__ A,
                                                 const u16* __restrict__ W,
                                                 float* __restrict__ C,
                                                 int M, int N, int K) {
  __shared__ u16 lds_a[2][128 * LDT];   // 20 KB
  __shared__ u16 lds_w[2][256 * LDT];   // 40 KB

  const int nblk = blockIdx.x * 256;
  const int mblk = blockIdx.y * 128;
  const int wave = threadIdx.x >> 5;
  const int lane = threadIdx.x & 31;
  const int l15  = lane & 15;
  const int hi   = lane >> 4;
  const int wm   = (wave >> 2) * 64;    // 0,64
  const int wn   = (wave & 3) * 64;     // 0,64,128,192

  // Per-tile staging: A 128x32 (2 chunks/thread), W 256x32 (4 chunks/thread)
  auto stage = [&](int buf, int k0) {
#pragma unroll
    for (int r = 0; r < 2; ++r) {
      int lin = threadIdx.x + r * 256;        // 0..511
      int row = lin >> 2;
      int qc  = (lin & 3) * 8;
      cp16_g2l(&lds_a[buf][row * LDT + qc],
               &A[(size_t)(mblk + row) * K + k0 + qc]);
    }
#pragma unroll
    for (int r = 0; r < 4; ++r) {
      int lin = threadIdx.x + r * 256;        // 0..1023
      int row = lin >> 2;
      int qc  = (lin & 3) * 8;
      cp16_g2l(&lds_w[buf][row * LDT + qc],
               &W[(size_t)(nblk + row) * K + k0 + qc]);
    }
  };

  v8f acc[4][4] = {};

  stage(0, 0);
  const int nsteps = K >> 5;
  for (int i = 0; i < nsteps; ++i) {
    const int buf = i & 1;
    if (i + 1 < nsteps) {
      stage(buf ^ 1, (i + 1) * 32);
      wait_async<6>();                  // 6 outstanding = next tile in flight
    } else {
      wait_async<0>();
    }
    __syncthreads();                    // tile i visible to all waves

    // A fragments: lane = row (M), k chunks {koff..koff+7, koff+16..koff+23}
    Frag16 af[4];
    const int koff = hi * 8;
#pragma unroll
    for (int t = 0; t < 4; ++t) {
      const u16* p = &lds_a[buf][(wm + t * 16 + l15) * LDT];
      af[t].q[0] = *(const v4u*)(p + koff);
      af[t].q[1] = *(const v4u*)(p + koff + 16);
    }
    // B fragments: lane = col (N), k chunks {8*hi..8*hi+7, 8*hi+16..8*hi+23}
    Frag16 bw[4];
#pragma unroll
    for (int j = 0; j < 4; ++j) {
      const u16* p = &lds_w[buf][(wn + j * 16 + l15) * LDT + hi * 8];
      bw[j].q[0] = *(const v4u*)(p);
      bw[j].q[1] = *(const v4u*)(p + 16);
    }
#pragma unroll
    for (int t = 0; t < 4; ++t)
#pragma unroll
      for (int j = 0; j < 4; ++j)
        acc[t][j] = wmma_bf16(af[t].v, bw[j].v, acc[t][j]);

    __syncthreads();                    // everyone done reading buf before reuse
  }

  // C layout: VGPR r -> row r + 8*hi, col = lane&15
#pragma unroll
  for (int t = 0; t < 4; ++t)
#pragma unroll
    for (int j = 0; j < 4; ++j)
#pragma unroll
      for (int r = 0; r < 8; ++r) {
        int m = mblk + wm + t * 16 + r + 8 * hi;
        int n = nblk + wn + j * 16 + l15;
        C[(size_t)m * N + n] = acc[t][j][r];
      }
}

// ---------- RoPE epilogue: f32 [b,s,h*dh] -> bf16 [b,h,s,dh] ----------
__global__ __launch_bounds__(256) void k_rope(const float* __restrict__ X,
                                              const int* __restrict__ pos,
                                              u16* __restrict__ Out) {
  int idx = blockIdx.x * 256 + threadIdx.x;            // pair index
  if (idx >= Bc * Sc * (Dc / 2)) return;
  int pr = idx & (Dc / 2 - 1);                         // 0..511
  int m  = idx / (Dc / 2);                             // 0..4095
  int s  = m & (Sc - 1);
  int b  = m / Sc;
  int h  = pr >> 5;                                    // head (32 pairs/head)
  int ii = pr & 31;                                    // freq index
  float invf = __expf(-(float)ii * 0.2878231366242557f);  // ln(1e4)/32
  float ang  = (float)pos[s] * invf;
  float sn, cs;
  __sincosf(ang, &sn, &cs);
  size_t ib = (size_t)m * Dc + h * DHc + 2 * ii;
  float x1 = X[ib], x2 = X[ib + 1];
  size_t ob = ((size_t)((b * Hc + h) * Sc + s)) * DHc + 2 * ii;
  Out[ob]     = f2b(x1 * cs - x2 * sn);
  Out[ob + 1] = f2b(x1 * sn + x2 * cs);
}

// ---------- V: f32 [b,s,h*dh] -> bf16 [b,h,dh,s] (transposed per head) ----------
__global__ __launch_bounds__(256) void k_vt(const float* __restrict__ Vf,
                                            u16* __restrict__ Vt) {
  int idx = blockIdx.x * 256 + threadIdx.x;
  if (idx >= Bc * Sc * Dc) return;
  int n = idx & (Dc - 1);
  int m = idx / Dc;
  int s = m & (Sc - 1);
  int b = m / Sc;
  int h = n >> 6, d = n & 63;
  Vt[((size_t)((b * Hc + h) * DHc + d)) * Sc + s] = f2b(Vf[idx]);
}

// ---------- causal flash attention ----------
// Q,K: bf16 [b,h,s,dh], Vt: bf16 [b,h,dh,s] -> Out bf16 [b*s, h*dh]
// Block = 8 waves x 16 query rows = 128 queries per (b,h) block. Keys in blocks of 32.
__global__ __launch_bounds__(256) void k_attn(const u16* __restrict__ Q,
                                              const u16* __restrict__ K,
                                              const u16* __restrict__ Vt,
                                              u16* __restrict__ Out) {
  __shared__ u16 lds_p[8][16 * LDT];     // per-wave P transpose slab

  const int qblk = blockIdx.x;           // 0..15
  const int h    = blockIdx.y;
  const int b    = blockIdx.z;
  const int wave = threadIdx.x >> 5;
  const int lane = threadIdx.x & 31;
  const int l15  = lane & 15;
  const int hi   = lane >> 4;

  const size_t head = (size_t)(b * Hc + h) * Sc * DHc;
  const u16* Qh = Q + head;              // [S,64]
  const u16* Kh = K + head;              // [S,64]
  const u16* Vh = Vt + head;             // [64,S]

  const int q0 = qblk * 128 + wave * 16;

  // Q A-fragments for dh chunks [0,32) and [32,64)
  Frag16 aq0, aq1;
  {
    const u16* qr = Qh + (size_t)(q0 + l15) * DHc;
    int koff = hi * 8;
    aq0.q[0] = *(const v4u*)(qr + koff);
    aq0.q[1] = *(const v4u*)(qr + koff + 16);
    aq1.q[0] = *(const v4u*)(qr + 32 + koff);
    aq1.q[1] = *(const v4u*)(qr + 32 + koff + 16);
  }

  v8f   o[4] = {};
  float mrow[8], lrow[8];
#pragma unroll
  for (int r = 0; r < 8; ++r) { mrow[r] = -1e30f; lrow[r] = 0.0f; }

  u16* pslab = &lds_p[wave][0];
  const int nkb = ((q0 + 15) >> 5) + 1;        // 32-key blocks up to the diagonal

  for (int kb = 0; kb < nkb; ++kb) {
    const int key0 = kb * 32;

    // prefetch next key block (global_prefetch_b8)
    if (kb + 1 < nkb) {
      __builtin_prefetch(Kh + (size_t)(key0 + 32 + l15) * DHc, 0, 1);
      __builtin_prefetch(Vh + (size_t)(l15)*Sc + key0 + 32, 0, 1);
    }

    // ---- scores: S = Q * K^T (two 16-key column tiles, dh=64 contraction) ----
    v8f s0 = {}, s1 = {};
    {
      Frag16 bk;
      const u16* kr = Kh + (size_t)(key0 + l15) * DHc + hi * 8;
      bk.q[0] = *(const v4u*)(kr);      bk.q[1] = *(const v4u*)(kr + 16);
      s0 = wmma_bf16(aq0.v, bk.v, s0);
      bk.q[0] = *(const v4u*)(kr + 32); bk.q[1] = *(const v4u*)(kr + 48);
      s0 = wmma_bf16(aq1.v, bk.v, s0);
    }
    {
      Frag16 bk;
      const u16* kr = Kh + (size_t)(key0 + 16 + l15) * DHc + hi * 8;
      bk.q[0] = *(const v4u*)(kr);      bk.q[1] = *(const v4u*)(kr + 16);
      s1 = wmma_bf16(aq0.v, bk.v, s1);
      bk.q[0] = *(const v4u*)(kr + 32); bk.q[1] = *(const v4u*)(kr + 48);
      s1 = wmma_bf16(aq1.v, bk.v, s1);
    }

    // ---- online softmax (per row r+8*hi of the 16-row tile) ----
#pragma unroll
    for (int r = 0; r < 8; ++r) {
      int row = q0 + r + 8 * hi;
      int c0 = key0 + l15, c1 = c0 + 16;
      float x0 = (c0 <= row) ? s0[r] * 0.125f : -1e30f;   // dh^-0.5 = 0.125
      float x1 = (c1 <= row) ? s1[r] * 0.125f : -1e30f;
      float rmax = half_max(fmaxf(x0, x1));
      float mnew = fmaxf(mrow[r], rmax);
      float p0 = __expf(x0 - mnew);
      float p1 = __expf(x1 - mnew);
      float rs = half_sum(p0 + p1);
      float corr = __expf(mrow[r] - mnew);
      lrow[r] = lrow[r] * corr + rs;
      mrow[r] = mnew;
      o[0][r] *= corr; o[1][r] *= corr; o[2][r] *= corr; o[3][r] *= corr;
      // transpose P through per-wave LDS (row-major 16x32, stride LDT)
      pslab[(r + 8 * hi) * LDT + l15]      = f2b(p0);
      pslab[(r + 8 * hi) * LDT + l15 + 16] = f2b(p1);
    }

    // ---- P A-fragment (16 queries x 32 keys) ----
    Frag16 pf;
    {
      const u16* pr = pslab + l15 * LDT;
      int koff = hi * 8;
      pf.q[0] = *(const v4u*)(pr + koff);
      pf.q[1] = *(const v4u*)(pr + koff + 16);
    }

    // ---- O += P * V  (B operand from transposed V: contiguous keys) ----
#pragma unroll
    for (int nt = 0; nt < 4; ++nt) {
      Frag16 vf;
      const u16* vr = Vh + (size_t)(nt * 16 + l15) * Sc + key0 + hi * 8;
      vf.q[0] = *(const v4u*)(vr);
      vf.q[1] = *(const v4u*)(vr + 16);
      o[nt] = wmma_bf16(pf.v, vf.v, o[nt]);
    }
  }

  // ---- epilogue: O/l -> bf16 [b*S+s, h*64+d] ----
#pragma unroll
  for (int r = 0; r < 8; ++r) {
    int row = q0 + r + 8 * hi;
    float inv = 1.0f / lrow[r];
    size_t ob = ((size_t)(b * Sc + row)) * Dc + h * DHc + l15;
#pragma unroll
    for (int nt = 0; nt < 4; ++nt)
      Out[ob + nt * 16] = f2b(o[nt][r] * inv);
  }
}

// ---------- host-side launch ----------
extern "C" void kernel_launch(void* const* d_in, const int* in_sizes, int n_in,
                              void* d_out, int out_size, void* d_ws, size_t ws_size,
                              hipStream_t stream) {
  const float* x  = (const float*)d_in[0];
  const float* WQ = (const float*)d_in[1];
  const float* WK = (const float*)d_in[2];
  const float* WV = (const float*)d_in[3];
  const float* WO = (const float*)d_in[4];
  const int*   tp = (const int*)d_in[5];
  float* out = (float*)d_out;

  char* ws = (char*)d_ws;
  u16*   xb  = (u16*)(ws + OFF_XB);
  u16*   wqb = (u16*)(ws + OFF_WQ);
  u16*   wkb = (u16*)(ws + OFF_WK);
  u16*   wvb = (u16*)(ws + OFF_WV);
  u16*   wob = (u16*)(ws + OFF_WO);
  float* qf  = (float*)(ws + OFF_QF);
  float* kf  = (float*)(ws + OFF_KF);
  float* vf  = (float*)(ws + OFF_VF);
  u16*   qb  = (u16*)(ws + OFF_QB);
  u16*   kb  = (u16*)(ws + OFF_KB);
  u16*   vt  = (u16*)(ws + OFF_VT);
  u16*   ab  = (u16*)(ws + OFF_AB);

  const int nx = Mtot * Dc;      // 4,194,304
  const int nw = Dc * Dc;        // 1,048,576

  // 1) convert inputs to bf16
  k_cvt<<<nx / 256, 256, 0, stream>>>(x, xb, nx);
  k_cvt<<<nw / 256, 256, 0, stream>>>(WQ, wqb, nw);
  k_cvt<<<nw / 256, 256, 0, stream>>>(WK, wkb, nw);
  k_cvt<<<nw / 256, 256, 0, stream>>>(WV, wvb, nw);
  k_cvt<<<nw / 256, 256, 0, stream>>>(WO, wob, nw);

  // 2) Q/K/V projections (NT GEMM, fp32 accumulate)
  dim3 gg(Dc / 256, Mtot / 128);           // (4, 32)
  k_gemm_nt<<<gg, 256, 0, stream>>>(xb, wqb, qf, Mtot, Dc, Dc);
  k_gemm_nt<<<gg, 256, 0, stream>>>(xb, wkb, kf, Mtot, Dc, Dc);
  k_gemm_nt<<<gg, 256, 0, stream>>>(xb, wvb, vf, Mtot, Dc, Dc);

  // 3) RoPE + head-major relayout; V transpose
  const int np = Bc * Sc * (Dc / 2);       // 2,097,152
  k_rope<<<np / 256, 256, 0, stream>>>(qf, tp, qb);
  k_rope<<<np / 256, 256, 0, stream>>>(kf, tp, kb);
  k_vt<<<nx / 256, 256, 0, stream>>>(vf, vt);

  // 4) causal flash attention
  dim3 ga(Sc / 128, Hc, Bc);               // (16, 16, 2)
  k_attn<<<ga, 256, 0, stream>>>(qb, kb, vt, ab);

  // 5) output projection straight into d_out (fp32)
  k_gemm_nt<<<gg, 256, 0, stream>>>(ab, wob, out, Mtot, Dc, Dc);
}
